// CharRNN_3753801417608
// MI455X (gfx1250) — matrix operand order repaired and checked by reading, not compiled, not run on hardware
//
#include <hip/hip_runtime.h>

// ---------------------------------------------------------------------------
// CharRNN (LSTM encoder + autoregressive LSTMCell decoder + linear head)
// CDNA5 / gfx1250: persistent scans, weights staged in LDS, f16 WMMA 16x16x32.
// ---------------------------------------------------------------------------

typedef _Float16 half_t;
typedef __attribute__((ext_vector_type(16))) _Float16 v16h;
typedef __attribute__((ext_vector_type(8)))  float    v8f;

#define B_    64
#define T_    1024
#define K_    512
#define KT_   16       // K/32 k-tiles per GEMM
#define NT_   32       // 512/16 n-tiles per gate
#define GRID_ 32       // persistent grid: 32 blocks x 128 threads (4 waves)

__device__ __forceinline__ float sigf(float x) { return 1.0f / (1.0f + __expf(-x)); }

union HU { half_t h; unsigned short u; };
struct Frag { uint4 a, b; };
__device__ __forceinline__ v16h fragH(const Frag& f) {
    union { uint4 u[2]; v16h h; } x; x.u[0] = f.a; x.u[1] = f.b; return x.h;
}

// ---------------------------------------------------------------------------
// Grid-wide split barrier (co-resident workgroups; bar[0]=arrive, bar[1]=gen)
// ---------------------------------------------------------------------------
__device__ __forceinline__ void grid_barrier(unsigned int* bar) {
    __syncthreads();
    if (threadIdx.x == 0) {
        __threadfence();   // release h-stores to device scope
        unsigned int gen = __hip_atomic_load(&bar[1], __ATOMIC_RELAXED, __HIP_MEMORY_SCOPE_AGENT);
        unsigned int arrived =
            __hip_atomic_fetch_add(&bar[0], 1u, __ATOMIC_ACQ_REL, __HIP_MEMORY_SCOPE_AGENT);
        if (arrived == gridDim.x - 1) {
            __hip_atomic_store(&bar[0], 0u, __ATOMIC_RELAXED, __HIP_MEMORY_SCOPE_AGENT);
            __hip_atomic_fetch_add(&bar[1], 1u, __ATOMIC_RELEASE, __HIP_MEMORY_SCOPE_AGENT);
        } else {
            while (__hip_atomic_load(&bar[1], __ATOMIC_ACQUIRE, __HIP_MEMORY_SCOPE_AGENT) == gen) {
                __builtin_amdgcn_s_sleep(1);
            }
        }
        __threadfence();   // acquire
    }
    __syncthreads();
}

// ---------------------------------------------------------------------------
// Pack [2048,512] f32 weight (optionally + a second) into a contiguous-k B-fragment
// layout: entry(ntg,kt,lane) = 8 dwords; lane%16 = col n, lane/16 = K half,
// dword r holds K = kt*32 + (lane/16)*16 + 2r, +1   (B[k][n] = W[n][k]).
// ---------------------------------------------------------------------------
__global__ __launch_bounds__(256) void pack_w(const float* __restrict__ W,
                                              const float* __restrict__ W2,
                                              unsigned int* __restrict__ out) {
    int id   = blockIdx.x * 256 + threadIdx.x;        // 0 .. 524287
    int r    = id & 7;
    int lane = (id >> 3) & 31;
    int kt   = (id >> 8) & 15;
    int ntg  = id >> 12;                              // 0..127
    int n  = ntg * 16 + (lane & 15);
    int hi = lane >> 4;
    int k  = kt * 32 + hi * 16 + 2 * r;
    float a0 = W[n * K_ + k];
    float a1 = W[n * K_ + k + 1];
    if (W2) { a0 += W2[n * K_ + k]; a1 += W2[n * K_ + k + 1]; }
    HU u0, u1; u0.h = (half_t)a0; u1.h = (half_t)a1;
    out[id] = (unsigned int)u0.u | ((unsigned int)u1.u << 16);
}

// Zero h ping buffer + barrier state (graph replay must be idempotent).
__global__ __launch_bounds__(256) void init_state(unsigned int* __restrict__ h0u,
                                                  unsigned int* __restrict__ bar) {
    int i = blockIdx.x * 256 + threadIdx.x;           // grid covers 16384
    if (i < (B_ * K_) / 2) h0u[i] = 0u;
    if (i < 2) bar[i] = 0u;
}

// code (f32, written by encoder) -> f16 A operand for decoder step 0.
__global__ __launch_bounds__(256) void prep_decoder(const float* __restrict__ code,
                                                    half_t* __restrict__ cf) {
    int i = blockIdx.x * 256 + threadIdx.x;           // 32768
    cf[i] = (half_t)code[i];
}

// ---------------------------------------------------------------------------
// Persistent LSTM scan.
//   * block = one n-tile (of 32) x all 4 m-tiles; 4 waves, 128 threads.
//   * The block's 64 KB of gate weights (4 gates x 16 k-tiles, B-fragment
//     layout) are staged in static LDS once; inner loop uses ds_load_b128.
//   * c (and encoder's code) live in registers for the whole scan; only h
//     round-trips through global, guarded by the grid barrier.
//   * LDS index is laundered through an opaque SGPR each step so LICM cannot
//     hoist the (invariant) fragment loads out of the time loop and spill.
//   MODE 0: encoder (x rank-1 term, code accumulation)
//   MODE 1: decoder (t=0: codef16 @ dec_Wih; t>=1: h @ (Wih+Whh); partials
//           of h.lin_W written per step for a deterministic reduce)
// ---------------------------------------------------------------------------
template <int MODE>
__global__ __launch_bounds__(128) void lstm_scan(
    const unsigned int* __restrict__ packW,   // main weights [128][16][32][8] dw
    const unsigned int* __restrict__ packW0,  // MODE 1: dec_Wih (t=0 only)
    const half_t*       __restrict__ cf,      // MODE 1: codef16 (A for t=0)
    half_t*             __restrict__ hA,      // ping (read at t=0)
    half_t*             __restrict__ hB,      // pong (written at t=0)
    const float*        __restrict__ bias,    // [2048]
    const float*        __restrict__ x,       // [64][1024]  (MODE 0)
    const float*        __restrict__ Wih,     // [2048]      (MODE 0)
    float*              __restrict__ code,    // [64][512]   (MODE 0, d_out)
    const float*        __restrict__ linW,    // [512]       (MODE 1)
    float*              __restrict__ part,    // [1024][64][32] (MODE 1)
    unsigned int*       __restrict__ bar) {
    __shared__ uint4 ldsW[4096];                             // 64 KB weights

    const int lane  = threadIdx.x & 31;
    const int ww    = threadIdx.x >> 5;                      // wave = m-tile
    const int ntile = blockIdx.x;                            // 0..31
    const int hi    = lane >> 4;
    const int coll  = lane & 15;
    const int m0    = ww * 16;
    const int cg    = ntile * 16 + coll;                     // gate-relative col

    // Stage this block's weights into LDS (4096 uint4, 128 threads).
    auto stageW = [&](const unsigned int* pwsrc) {
        const uint4* pw4 = (const uint4*)pwsrc;
        for (int i = threadIdx.x; i < 4096; i += 128) {
            int half = i & 1;
            int e    = i >> 1;
            int ln   = e & 31;
            int kt   = (e >> 5) & 15;
            int g    = e >> 9;                               // 0..3
            ldsW[i] = pw4[(((g * 32 + ntile) * 16 + kt) * 32 + ln) * 2 + half];
        }
    };

    // Loop-invariant scalars.
    const float bI = bias[cg], bF = bias[cg + 512], bG = bias[cg + 1024], bO = bias[cg + 1536];
    float wI = 0.f, wF = 0.f, wG = 0.f, wO = 0.f, lw = 0.f;
    if (MODE == 0) { wI = Wih[cg]; wF = Wih[cg + 512]; wG = Wih[cg + 1024]; wO = Wih[cg + 1536]; }
    else           { lw = linW[cg]; }

    // Register-resident cell state / code accumulator: this lane always owns
    // elements (r = m0 + v + 8*hi, col = cg) for v = 0..7.
    float creg[8], codeReg[8];
#pragma unroll
    for (int v = 0; v < 8; ++v) { creg[v] = 0.f; codeReg[v] = 0.f; }

    const size_t aLaneOff = (size_t)(m0 + coll) * (K_ * 2) + hi * 32;

#define LOADA(ABASE, KTI, FA)                                                  \
    do {                                                                       \
        const uint4* ap_ = (const uint4*)((ABASE) + (KTI) * 64);               \
        FA.a = ap_[0]; FA.b = ap_[1];                                          \
    } while (0)
#define LOADB(KTI, LB, F0, F1, F2, F3)                                         \
    do {                                                                       \
        int o_ = (KTI) * 64 + (LB);                                            \
        F0.a = ldsW[o_];        F0.b = ldsW[o_ + 1];                           \
        F1.a = ldsW[o_ + 1024]; F1.b = ldsW[o_ + 1025];                        \
        F2.a = ldsW[o_ + 2048]; F2.b = ldsW[o_ + 2049];                        \
        F3.a = ldsW[o_ + 3072]; F3.b = ldsW[o_ + 3073];                        \
    } while (0)

    // One full LSTM time step (GEMM from LDS weights + fused cell update).
    auto do_step = [&](int t, const half_t* Ain, half_t* Aout) {
        // Opaque zero regenerated every step: blocks LICM from hoisting the
        // loop-invariant LDS fragment loads (which would spill to scratch).
        unsigned lz;
        asm volatile("s_mov_b32 %0, 0" : "=s"(lz));
        const int lb = lane * 2 + (int)lz;

        const char* aBase = (const char*)Ain + aLaneOff;

        v8f accI{}, accF{}, accG{}, accO{};
        Frag A0, A1;
        LOADA(aBase, 0, A0);
#pragma unroll
        for (int kt = 0; kt < KT_; ++kt) {
            Frag b0, b1, b2, b3;
            LOADB(kt, lb, b0, b1, b2, b3);
            if (kt + 1 < KT_) LOADA(aBase, kt + 1, A1);
            accI = __builtin_amdgcn_wmma_f32_16x16x32_f16(false, fragH(A0), false, fragH(b0), (short)0, accI, false, false);
            accF = __builtin_amdgcn_wmma_f32_16x16x32_f16(false, fragH(A0), false, fragH(b1), (short)0, accF, false, false);
            accG = __builtin_amdgcn_wmma_f32_16x16x32_f16(false, fragH(A0), false, fragH(b2), (short)0, accG, false, false);
            accO = __builtin_amdgcn_wmma_f32_16x16x32_f16(false, fragH(A0), false, fragH(b3), (short)0, accO, false, false);
            A0 = A1;
        }

        // Epilogue: C layout -> VGPR v = row m0+v+8*hi, lane = col cg.
        float lp[8];
#pragma unroll
        for (int v = 0; v < 8; ++v) {
            const int r = m0 + v + 8 * hi;
            float gi = accI[v] + bI;
            float gf = accF[v] + bF;
            float gg = accG[v] + bG;
            float go = accO[v] + bO;
            if (MODE == 0) {
                const float xv = x[r * T_ + t];              // rank-1 input term
                gi += xv * wI; gf += xv * wF; gg += xv * wG; go += xv * wO;
            }
            const float cn = sigf(gf) * creg[v] + sigf(gi) * tanhf(gg);
            const float hn = sigf(go) * tanhf(cn);
            creg[v] = cn;
            Aout[r * K_ + cg] = (half_t)hn;
            if (MODE == 0) codeReg[v] += hn;
            else           lp[v] = hn * lw;
        }

        if (MODE == 1) {
            // Reduce the 16 columns of each row within the 16-lane half.
#pragma unroll
            for (int v = 0; v < 8; ++v) {
                float p = lp[v];
                p += __shfl_xor(p, 1, 16);
                p += __shfl_xor(p, 2, 16);
                p += __shfl_xor(p, 4, 16);
                p += __shfl_xor(p, 8, 16);
                if (coll == 0) {
                    const int r = m0 + v + 8 * hi;
                    part[(size_t)t * (B_ * NT_) + r * NT_ + ntile] = p;
                }
            }
        }
    };

    // ---- t = 0 (decoder uses codef16 @ dec_Wih; encoder uses zeroed hA) ----
    stageW(MODE == 1 ? packW0 : packW);
    __syncthreads();
    do_step(0, (MODE == 1) ? cf : hA, hB);
    grid_barrier(bar);

    if (MODE == 1) {                       // switch to (dec_Wih + dec_Whh)
        stageW(packW);
        __syncthreads();
    }

    // ---- t = 1 .. T-1 ----
    for (int t = 1; t < T_; ++t) {
        const half_t* Ain  = (t & 1) ? hB : hA;
        half_t*       Aout = (t & 1) ? hA : hB;
        do_step(t, Ain, Aout);
        grid_barrier(bar);
    }

    if (MODE == 0) {
#pragma unroll
        for (int v = 0; v < 8; ++v) code[(m0 + v + 8 * hi) * K_ + cg] = codeReg[v];
    }
#undef LOADA
#undef LOADB
}

// out[b*T+t] = lin_b + sum of 32 n-tile partials (fixed order -> deterministic)
__global__ __launch_bounds__(256) void reduce_out(const float* __restrict__ part,
                                                  const float* __restrict__ lin_b,
                                                  float* __restrict__ out) {
    int id = blockIdx.x * 256 + threadIdx.x;              // 65536 = T*B
    int t = id >> 6;
    int b = id & 63;
    const float* p = part + (size_t)t * (B_ * NT_) + b * NT_;
    float s = lin_b[0];
#pragma unroll
    for (int n = 0; n < NT_; ++n) s += p[n];
    out[b * T_ + t] = s;
}

// ---------------------------------------------------------------------------
extern "C" void kernel_launch(void* const* d_in, const int* in_sizes, int n_in,
                              void* d_out, int out_size, void* d_ws, size_t ws_size,
                              hipStream_t stream) {
    const float* x        = (const float*)d_in[0];
    const float* enc_Wih  = (const float*)d_in[1];
    const float* enc_Whh  = (const float*)d_in[2];
    const float* enc_b    = (const float*)d_in[3];
    const float* dec_Wih  = (const float*)d_in[4];
    const float* dec_Whh  = (const float*)d_in[5];
    const float* dec_b    = (const float*)d_in[6];
    const float* lin_W    = (const float*)d_in[7];
    const float* lin_b    = (const float*)d_in[8];

    float* out  = (float*)d_out;            // [B*T]
    float* code = out + B_ * T_;            // [B*K]

    char* ws = (char*)d_ws;
    const size_t PW = 524288u * 4u;                         // 2 MB per matrix
    unsigned int* pwE  = (unsigned int*)(ws);
    unsigned int* pwDI = (unsigned int*)(ws + PW);
    unsigned int* pwDS = (unsigned int*)(ws + 2 * PW);
    half_t* h0 = (half_t*)(ws + 3 * PW);                    // 64 KB
    half_t* h1 = h0 + B_ * K_;                              // 64 KB
    half_t* cf = h1 + B_ * K_;                              // 64 KB
    unsigned int* bar = (unsigned int*)(ws + 3 * PW + 3u * 65536u);      // 8 B
    float* part = (float*)(ws + 3 * PW + 3u * 65536u + 256u);            // 8 MB

    init_state<<<64, 256, 0, stream>>>((unsigned int*)h0, bar);
    pack_w<<<2048, 256, 0, stream>>>(enc_Whh, nullptr, pwE);
    pack_w<<<2048, 256, 0, stream>>>(dec_Wih, nullptr, pwDI);
    pack_w<<<2048, 256, 0, stream>>>(dec_Wih, dec_Whh, pwDS);  // inp==h, t>=1

    // Encoder: persistent scan, reads h0 first (zeroed).
    lstm_scan<0><<<GRID_, 128, 0, stream>>>(pwE, pwE, nullptr, h0, h1,
                                            enc_b, x, enc_Wih, code,
                                            nullptr, nullptr, bar);

    prep_decoder<<<128, 256, 0, stream>>>(code, cf);

    // Decoder: t=0 codef16 @ dec_Wih -> h0; then ping-pong with (Wih+Whh).
    lstm_scan<1><<<GRID_, 128, 0, stream>>>(pwDS, pwDI, cf, /*hA=*/h1, /*hB=*/h0,
                                            dec_b, nullptr, nullptr, nullptr,
                                            lin_W, part, bar);

    // Deterministic final reduction for the linear head.
    reduce_out<<<256, 256, 0, stream>>>(part, lin_b, out);
}
